// EntropyRecyclingLanguageNet_11338713661903
// MI455X (gfx1250) — matrix operand order repaired and checked by reading, not compiled, not run on hardware
//
#include <hip/hip_runtime.h>
#include <hip/hip_bf16.h>
#include <stdint.h>

// Problem constants (match reference)
#define B_ 8192
#define D_ 128
#define P_ 64
#define V_ 32000

// ---------------------------------------------------------------- types
typedef __attribute__((ext_vector_type(16))) __bf16      v16bf;
typedef __attribute__((ext_vector_type(8)))  float        v8f;
typedef __attribute__((ext_vector_type(4)))  unsigned int v4u;

union Frag {          // 32 bytes: one WMMA 16-bit A or B operand (8 VGPRs)
    v16bf v;
    v4u   q[2];
};

__device__ __forceinline__ uint16_t f2bf_bits(float f) {
    union { float f; uint32_t u; } v; v.f = f;
    uint32_t r = v.u + 0x7FFFu + ((v.u >> 16) & 1u);   // round-to-nearest-even
    return (uint16_t)(r >> 16);
}
__device__ __forceinline__ __bf16 f2bf(float f) {
    uint16_t b = f2bf_bits(f);
    __bf16 h; __builtin_memcpy(&h, &b, 2);
    return h;
}

// ---------------------------------------------------------------- kernel 1
// Repack out_w [D_,V_] f32 row-major into bf16 B-fragments.
// Fragment layout per (nTile, kk) — contiguous k order, same as the A fragments:
//   lane 0..15  : N = lane,   K = kk*32 + 0..15  (8 packed dwords, K pairs)
//   lane 16..31 : N = lane-16,K = kk*32 + 16..31
// Flat dword index = (((nTile*4 + kk)*32 + lane)*8 + dw)
__global__ void pack_b_kernel(const float* __restrict__ W,
                              uint32_t* __restrict__ Bp) {
    int idx   = blockIdx.x * 256 + threadIdx.x;      // 0 .. 2,047,999
    int dw    = idx & 7;
    int lane  = (idx >> 3) & 31;
    int kk    = (idx >> 8) & 3;
    int nTile = idx >> 10;
    int k0 = kk * 32 + ((lane >> 4) << 4) + dw * 2;
    int n  = nTile * 16 + (lane & 15);
    float f0 = W[(size_t)k0 * V_ + n];
    float f1 = W[(size_t)(k0 + 1) * V_ + n];
    Bp[idx] = (uint32_t)f2bf_bits(f0) | ((uint32_t)f2bf_bits(f1) << 16);
}

// ---------------------------------------------------------------- kernel 2
// One 128-thread block per batch row: attention logits -> softmax ->
// reconstruction -> self-predict -> paradox gate -> gated (bf16).
__global__ void fused_rows_kernel(const float* __restrict__ x,
                                  const float* __restrict__ pattern_dict,
                                  const float* __restrict__ attn_w,
                                  const float* __restrict__ attn_b,
                                  const float* __restrict__ self_w,
                                  const float* __restrict__ self_b,
                                  __bf16* __restrict__ gatedB) {
    __shared__ float sx[D_];
    __shared__ float sw[P_];
    __shared__ float srec[D_];
    __shared__ float sred[D_];

    const int b = blockIdx.x;
    const int t = threadIdx.x;

    sx[t] = x[(size_t)b * D_ + t];
    __syncthreads();

    // logits over P_=64 patterns (threads 0..63)
    float logit = 0.0f;
    if (t < P_) {
        float acc = attn_b[t];
        #pragma unroll 8
        for (int d = 0; d < D_; ++d)
            acc = __builtin_fmaf(sx[d], attn_w[d * P_ + t], acc);
        logit = acc;
    }

    // softmax max
    sred[t] = (t < P_) ? logit : -1e30f;
    __syncthreads();
    for (int s = 32; s > 0; s >>= 1) {
        if (t < s) sred[t] = fmaxf(sred[t], sred[t + s]);
        __syncthreads();
    }
    float mx = sred[0];
    __syncthreads();

    // softmax sum
    float e = (t < P_) ? __expf(logit - mx) : 0.0f;
    sred[t] = e;
    __syncthreads();
    for (int s = 64; s > 0; s >>= 1) {
        if (t < s) sred[t] += sred[t + s];
        __syncthreads();
    }
    float inv_sum = 1.0f / sred[0];
    __syncthreads();
    if (t < P_) sw[t] = e * inv_sum;
    __syncthreads();

    // reconstruction rec[t] = sum_p w[p] * pattern_dict[p, t]
    float rec = 0.0f;
    #pragma unroll 8
    for (int p = 0; p < P_; ++p)
        rec = __builtin_fmaf(sw[p], pattern_dict[p * D_ + t], rec);
    srec[t] = rec;
    __syncthreads();

    // self predictor
    float sp = self_b[t];
    #pragma unroll 8
    for (int d = 0; d < D_; ++d)
        sp = __builtin_fmaf(srec[d], self_w[d * D_ + t], sp);

    // paradox magnitude (L2 over 128)
    float par = sp - rec;
    sred[t] = par * par;
    __syncthreads();
    for (int s = 64; s > 0; s >>= 1) {
        if (t < s) sred[t] += sred[t + s];
        __syncthreads();
    }
    float mag = __builtin_sqrtf(sred[0]);
    float sig = 1.0f / (1.0f + __expf(-mag));

    gatedB[(size_t)b * D_ + t] = f2bf(rec * sig);
}

// ---------------------------------------------------------------- kernel 3
// out[B_,V_] = gated(bf16) @ out_w(bf16,packed) + out_b, f32 accumulate.
// 128 threads = 4 waves; each wave owns one 16-row M tile x 4 N-tiles (64 cols).
// K = 128 unrolled as 4 x v_wmma_f32_16x16x32_bf16.
#define NT_PER_WAVE 4
__global__ void gemm_out_kernel(const __bf16* __restrict__ A,      // [B_,D_] row-major bf16
                                const uint32_t* __restrict__ Bp,   // packed B fragments
                                const float* __restrict__ bias,    // [V_]
                                float* __restrict__ out) {         // [B_,V_]
    const int wave = threadIdx.x >> 5;          // 0..3
    const int lane = threadIdx.x & 31;
    const int mBase  = blockIdx.y * 16;
    const int ntBase = (blockIdx.x * 4 + wave) * NT_PER_WAVE;   // 16 N-tiles / block

    // A-fragment addressing (contiguous k order, matching packed B):
    //   lane 0..15  : M = lane,    K {0..15}
    //   lane 16..31 : M = lane-16, K {16..31}
    const int m    = mBase + (lane & 15);
    const int kOff = (lane >> 4) * 16;

    v8f acc[NT_PER_WAVE] = {};

    #pragma unroll
    for (int kk = 0; kk < 4; ++kk) {
        Frag a;
        const __bf16* pa = A + (size_t)m * D_ + kk * 32 + kOff;
        a.q[0] = *(const v4u*)(pa);            // K 0..7   of this lane half
        a.q[1] = *(const v4u*)(pa + 8);        // K 8..15  of this lane half

        #pragma unroll
        for (int nt = 0; nt < NT_PER_WAVE; ++nt) {
            Frag bfr;
            const uint32_t* pb =
                Bp + (((size_t)(ntBase + nt) * 4 + kk) * 32 + lane) * 8;
            bfr.q[0] = *(const v4u*)(pb);
            bfr.q[1] = *(const v4u*)(pb + 4);
            acc[nt] = __builtin_amdgcn_wmma_f32_16x16x32_bf16(
                false, a.v, false, bfr.v, (short)0, acc[nt], false, false);
        }
    }

    // C/D layout: VGPR r -> (M = mBase + r [+8 for hi lanes], N = lane & 15)
    const int rowBase = mBase + ((lane >> 4) << 3);
    const int col     = lane & 15;
    #pragma unroll
    for (int nt = 0; nt < NT_PER_WAVE; ++nt) {
        int n = (ntBase + nt) * 16 + col;
        float bv = bias[n];
        float* po = out + (size_t)rowBase * V_ + n;
        #pragma unroll
        for (int r = 0; r < 8; ++r)
            po[(size_t)r * V_] = acc[nt][r] + bv;
    }
}

// ---------------------------------------------------------------- launch
extern "C" void kernel_launch(void* const* d_in, const int* in_sizes, int n_in,
                              void* d_out, int out_size, void* d_ws, size_t ws_size,
                              hipStream_t stream) {
    (void)in_sizes; (void)n_in; (void)out_size; (void)ws_size;

    const float* x            = (const float*)d_in[0];
    const float* pattern_dict = (const float*)d_in[1];
    const float* attn_w       = (const float*)d_in[2];
    const float* attn_b       = (const float*)d_in[3];
    const float* self_w       = (const float*)d_in[4];
    const float* self_b       = (const float*)d_in[5];
    const float* out_w        = (const float*)d_in[6];
    const float* out_b        = (const float*)d_in[7];
    float*       out          = (float*)d_out;

    char* ws = (char*)d_ws;
    __bf16*   gatedB = (__bf16*)ws;                              // 8192*128*2 = 2 MiB
    uint32_t* Bp     = (uint32_t*)(ws + (size_t)B_ * D_ * 2);    // 2000*4*32*8*4 = 8 MB

    // 1) repack out_w into bf16 WMMA B-fragments (2,048,000 dwords)
    pack_b_kernel<<<8000, 256, 0, stream>>>(out_w, Bp);

    // 2) fused per-row pipeline -> gated bf16 (independent of pack_b)
    fused_rows_kernel<<<B_, D_, 0, stream>>>(x, pattern_dict, attn_w, attn_b,
                                             self_w, self_b, gatedB);

    // 3) big vocab GEMM with bf16 WMMA, bias fused into the store
    dim3 grid(V_ / (16 * 4 * NT_PER_WAVE), B_ / 16);   // (125, 512)
    gemm_out_kernel<<<grid, 128, 0, stream>>>(gatedB, Bp, out_b, out);
}
